// MambaLayer_45414984188495
// MI455X (gfx1250) — compile-verified, matrix operand order checked by reading, not run on hardware
//
#include <hip/hip_runtime.h>
#include <hip/hip_bf16.h>
#include <math.h>

// ---------------- problem constants ----------------
#define BSZ      2
#define CDIM     64
#define TDIM     8
#define HDIM     64
#define WDIM     64
#define HF       32
#define WF       32
#define LF       (TDIM*HF*WF)      // 8192
#define HC       16
#define WC       16
#define LC       (TDIM*HC*WC)      // 2048
#define DINNER   128
#define DSTATE   16
#define DCONV    4
#define DTRANK   4
#define XPROJ_N  (DTRANK + 2*DSTATE)   // 36

typedef __attribute__((ext_vector_type(16))) _Float16 v16h;
typedef __attribute__((ext_vector_type(8)))  _Float16 v8h;
typedef __attribute__((ext_vector_type(8)))  float    v8f;

__device__ __forceinline__ float sigmoidf_(float x) { return 1.0f / (1.0f + __expf(-x)); }
__device__ __forceinline__ float softplusf_(float x) { return (x > 20.0f) ? x : log1pf(__expf(x)); }

// ---------------- pooling: (B,C,T,H,W) -> fine tokens (B, Lf, C) ----------------
__global__ void pool_fine_kernel(const float* __restrict__ x, float* __restrict__ tokf) {
    int idx = blockIdx.x * blockDim.x + threadIdx.x;          // (b, l, c)
    if (idx >= BSZ * LF * CDIM) return;
    int c = idx & 63;
    int l = (idx >> 6) & (LF - 1);
    int b = idx >> 19;
    int w = l & (WF - 1);
    int h = (l >> 5) & (HF - 1);
    int t = l >> 10;
    const float* xb = x + ((((size_t)b * CDIM + c) * TDIM + t) * HDIM + 2 * h) * WDIM + 2 * w;
    tokf[idx] = 0.25f * (xb[0] + xb[1] + xb[WDIM] + xb[WDIM + 1]);
}

// coarse tokens (B, Lc, C) = 2x2 mean of fine tokens
__global__ void pool_coarse_kernel(const float* __restrict__ tokf, float* __restrict__ tokc) {
    int idx = blockIdx.x * blockDim.x + threadIdx.x;
    if (idx >= BSZ * LC * CDIM) return;
    int c = idx & 63;
    int l = (idx >> 6) & (LC - 1);
    int b = idx >> 17;
    int w = l & (WC - 1);
    int h = (l >> 4) & (HC - 1);
    int t = l >> 8;
    int l00 = (t * HF + 2 * h) * WF + 2 * w;
    const float* tf = tokf + ((size_t)b * LF) * CDIM + c;
    float s = tf[(size_t)l00 * CDIM] + tf[(size_t)(l00 + 1) * CDIM]
            + tf[(size_t)(l00 + WF) * CDIM] + tf[(size_t)(l00 + WF + 1) * CDIM];
    tokc[idx] = 0.25f * s;
}

// ---------------- f32 -> f16 (weights) ----------------
__global__ void to_f16_kernel(const float* __restrict__ src, _Float16* __restrict__ dst, int n) {
    int i = blockIdx.x * blockDim.x + threadIdx.x;
    if (i < n) dst[i] = (_Float16)src[i];
}

// ---------------- layernorm over C=64 per token; f16 output (GEMM operand) ----------
__global__ void ln_kernel(const float* __restrict__ tok, _Float16* __restrict__ out,
                          const float* __restrict__ g, const float* __restrict__ bb, int ntok) {
    int t = blockIdx.x * blockDim.x + threadIdx.x;
    if (t >= ntok) return;
    const float* p = tok + (size_t)t * CDIM;
    float m = 0.f;
    #pragma unroll 8
    for (int c = 0; c < CDIM; ++c) m += p[c];
    m *= (1.0f / CDIM);
    float v = 0.f;
    #pragma unroll 8
    for (int c = 0; c < CDIM; ++c) { float d = p[c] - m; v += d * d; }
    v *= (1.0f / CDIM);
    float r = rsqrtf(v + 1e-5f);
    _Float16* o = out + (size_t)t * CDIM;
    #pragma unroll 8
    for (int c = 0; c < CDIM; ++c) o[c] = (_Float16)((p[c] - m) * r * g[c] + bb[c]);
}

// residual + layernorm: out = LN(tok + y), f32 output (branch result)
__global__ void res_ln_kernel(const float* __restrict__ tok, const float* __restrict__ y,
                              float* __restrict__ out,
                              const float* __restrict__ g, const float* __restrict__ bb, int ntok) {
    int t = blockIdx.x * blockDim.x + threadIdx.x;
    if (t >= ntok) return;
    const float* p = tok + (size_t)t * CDIM;
    const float* q = y   + (size_t)t * CDIM;
    float m = 0.f;
    #pragma unroll 8
    for (int c = 0; c < CDIM; ++c) m += p[c] + q[c];
    m *= (1.0f / CDIM);
    float v = 0.f;
    #pragma unroll 8
    for (int c = 0; c < CDIM; ++c) { float d = p[c] + q[c] - m; v += d * d; }
    v *= (1.0f / CDIM);
    float r = rsqrtf(v + 1e-5f);
    float* o = out + (size_t)t * CDIM;
    #pragma unroll 8
    for (int c = 0; c < CDIM; ++c) o[c] = (p[c] + q[c] - m) * r * g[c] + bb[c];
}

// ---------------- WMMA GEMM: Out(M,N) = A(M,K) * W(N,K)^T, f16 in, f32 out -----------
// One wave32 computes a 16 x 64 strip (4 accumulator tiles sharing the A fragment).
// Requirements (met here): M%16==0, N%64==0, K%32==0; A,W 16-byte aligned.
// A fragment (16-bit A 16x32): lane = m + 16*khalf holds f16 elems
//   [0..7]  = K = khalf*8 + j,  [8..15] = K = 16 + khalf*8 + j   (m = lane&15)
// B fragment (32x16) mirrors A with n in place of m (W stored (N,K) row-major).
// C/D f32 layout: lanes 0-15 -> N=lane, M=vgpr r; lanes 16-31 -> N=lane-16, M=8+r.
__device__ __forceinline__ v16h load_frag(const _Float16* row, int kh) {
    v8h lo = *(const v8h*)(row + kh);
    v8h hi = *(const v8h*)(row + 16 + kh);
    return __builtin_shufflevector(lo, hi, 0,1,2,3,4,5,6,7,8,9,10,11,12,13,14,15);
}

__global__ void gemm_wmma_f16_kernel(const _Float16* __restrict__ A,
                                     const _Float16* __restrict__ Wt,
                                     float* __restrict__ Out, int M, int N, int K) {
    int wave = threadIdx.x >> 5;
    int lane = threadIdx.x & 31;
    int ngrp = N >> 6;                             // groups of 4 N-tiles
    int tile = blockIdx.x * 4 + wave;
    int total = (M >> 4) * ngrp;
    if (tile >= total) return;
    int tm  = tile / ngrp;
    int tng = tile - tm * ngrp;
    int mr  = (tm << 4) + (lane & 15);
    int nb  = (tng << 6) + (lane & 15);            // base N row for fragment loads
    int kh  = (lane >> 4) << 3;                    // khalf*8

    v8f acc0 = {}, acc1 = {}, acc2 = {}, acc3 = {};
    for (int k0 = 0; k0 < K; k0 += 32) {
        const _Float16* Ar = A + (size_t)mr * K + k0;
        const _Float16* Wr = Wt + (size_t)nb * K + k0;
        v16h a  = load_frag(Ar, kh);
        v16h b0 = load_frag(Wr,                kh);
        v16h b1 = load_frag(Wr + (size_t)16*K, kh);
        v16h b2 = load_frag(Wr + (size_t)32*K, kh);
        v16h b3 = load_frag(Wr + (size_t)48*K, kh);
        acc0 = __builtin_amdgcn_wmma_f32_16x16x32_f16(false, a, false, b0, (short)0, acc0, false, false);
        acc1 = __builtin_amdgcn_wmma_f32_16x16x32_f16(false, a, false, b1, (short)0, acc1, false, false);
        acc2 = __builtin_amdgcn_wmma_f32_16x16x32_f16(false, a, false, b2, (short)0, acc2, false, false);
        acc3 = __builtin_amdgcn_wmma_f32_16x16x32_f16(false, a, false, b3, (short)0, acc3, false, false);
    }
    int rb = (tm << 4) + ((lane >> 4) << 3);
    int cn = (tng << 6) + (lane & 15);
    #pragma unroll
    for (int r = 0; r < 8; ++r) {
        float* o = Out + (size_t)(rb + r) * N + cn;
        o[0]  = acc0[r];
        o[16] = acc1[r];
        o[32] = acc2[r];
        o[48] = acc3[r];
    }
}

// ---------------- causal depthwise conv (k=4) + SiLU: xz[:, :128] -> u ----------------
__global__ void conv_silu_kernel(const float* __restrict__ xz, const float* __restrict__ cw,
                                 const float* __restrict__ cb, float* __restrict__ u, int L) {
    int idx = blockIdx.x * blockDim.x + threadIdx.x;        // (b, l, d)
    if (idx >= BSZ * L * DINNER) return;
    int d = idx & (DINNER - 1);
    int token = idx >> 7;                                   // b*L + l
    int b = token / L;
    int l = token - b * L;
    float acc = cb[d];
    #pragma unroll
    for (int j = 0; j < DCONV; ++j) {
        int ls = l - (DCONV - 1) + j;
        if (ls >= 0)
            acc += cw[d * DCONV + j] * xz[((size_t)(b * L + ls)) * 256 + d];
    }
    u[idx] = acc * sigmoidf_(acc);
}

// ---------------- x_dbl projection + dt softplus; one block (128 thr) per token -------
__global__ void xdbl_dt_kernel(const float* __restrict__ u, const float* __restrict__ xproj,
                               const float* __restrict__ dtw, const float* __restrict__ dtb,
                               float* __restrict__ dt, float* __restrict__ Bm,
                               float* __restrict__ Cm) {
    int token = blockIdx.x;
    int tid = threadIdx.x;
    __shared__ float urow[DINNER];
    __shared__ float xd[XPROJ_N];
    urow[tid] = u[(size_t)token * DINNER + tid];
    __syncthreads();
    if (tid < XPROJ_N) {
        float a = 0.f;
        #pragma unroll 8
        for (int k = 0; k < DINNER; ++k) a += xproj[tid * DINNER + k] * urow[k];
        xd[tid] = a;
    }
    __syncthreads();
    float s = dtb[tid];
    #pragma unroll
    for (int r = 0; r < DTRANK; ++r) s += xd[r] * dtw[tid * DTRANK + r];
    dt[(size_t)token * DINNER + tid] = softplusf_(s);
    if (tid < DSTATE)            Bm[(size_t)token * DSTATE + tid] = xd[DTRANK + tid];
    else if (tid < 2 * DSTATE)   Cm[(size_t)token * DSTATE + (tid - DSTATE)] = xd[DTRANK + DSTATE + (tid - DSTATE)];
}

// ---------------- selective scan: block = 16 d x 16 n lanes; sequential over L --------
#define SCAN_PF 8
__global__ void scan_kernel(const float* __restrict__ dt, const float* __restrict__ u,
                            const float* __restrict__ Bm, const float* __restrict__ Cm,
                            const float* __restrict__ A_log, float* __restrict__ ys, int L) {
    int n  = threadIdx.x & (DSTATE - 1);
    int dg = threadIdx.x >> 4;                 // 0..15
    int chunk = blockIdx.x & 7;                // 8 chunks of 16 d
    int b = blockIdx.x >> 3;
    int d = chunk * 16 + dg;
    float Av = -__expf(A_log[d * DSTATE + n]);
    float h = 0.f;
    size_t base = (size_t)b * L;
    for (int l = 0; l < L; ++l) {
        size_t tk = base + l;
        if (l + SCAN_PF < L) {                 // keep HBM requests in flight
            size_t tp = tk + SCAN_PF;
            __builtin_prefetch(&dt[tp * DINNER + d], 0, 1);
            __builtin_prefetch(&u [tp * DINNER + d], 0, 1);
            __builtin_prefetch(&Bm[tp * DSTATE + n], 0, 1);
        }
        float dtl = dt[tk * DINNER + d];
        float ul  = u [tk * DINNER + d];
        float Bl  = Bm[tk * DSTATE + n];
        float Cl  = Cm[tk * DSTATE + n];
        h = __expf(dtl * Av) * h + dtl * Bl * ul;
        float p = h * Cl;
        p += __shfl_xor(p, 8, 16);
        p += __shfl_xor(p, 4, 16);
        p += __shfl_xor(p, 2, 16);
        p += __shfl_xor(p, 1, 16);
        if (n == 0) ys[tk * DINNER + d] = p;
    }
}

// ---------------- gate: y16 = (ys + u*D) * silu(z)  (f16 output for out-proj) ---------
__global__ void gate_kernel(const float* __restrict__ ys, const float* __restrict__ u,
                            const float* __restrict__ Dp, const float* __restrict__ xz,
                            _Float16* __restrict__ y16, int total) {
    int idx = blockIdx.x * blockDim.x + threadIdx.x;        // (token, d)
    if (idx >= total) return;
    int d = idx & (DINNER - 1);
    int token = idx >> 7;
    float y = ys[idx] + u[idx] * Dp[d];
    float z = xz[(size_t)token * 256 + DINNER + d];
    y16[idx] = (_Float16)(y * z * sigmoidf_(z));
}

// ---------------- final: out_fine + bilinear-up(out_coarse) + 0.1*x_fine -------------
// output layout (B, C, T, 32, 32)
__global__ void final_kernel(const float* __restrict__ obrF, const float* __restrict__ obrC,
                             const float* __restrict__ tokf, float* __restrict__ out) {
    int idx = blockIdx.x * blockDim.x + threadIdx.x;
    if (idx >= BSZ * CDIM * TDIM * HF * WF) return;
    int w = idx & 31;
    int h = (idx >> 5) & 31;
    int t = (idx >> 10) & 7;
    int c = (idx >> 13) & 63;
    int b = idx >> 19;
    int lf = (t * HF + h) * WF + w;
    float vf = obrF[((size_t)b * LF + lf) * CDIM + c];
    float xf = tokf[((size_t)b * LF + lf) * CDIM + c];
    // half-pixel bilinear, scale 2 (T is identity)
    float hs = 0.5f * h - 0.25f, wsrc = 0.5f * w - 0.25f;
    int h0 = (int)floorf(hs); float fh = hs - (float)h0;
    int w0 = (int)floorf(wsrc); float fw = wsrc - (float)w0;
    int h0c = max(h0, 0), h1c = min(h0 + 1, HC - 1);
    int w0c = max(w0, 0), w1c = min(w0 + 1, WC - 1);
    const float* cb = obrC + (size_t)b * LC * CDIM + c;
    int lt = t * HC;
    float c00 = cb[(size_t)((lt + h0c) * WC + w0c) * CDIM];
    float c01 = cb[(size_t)((lt + h0c) * WC + w1c) * CDIM];
    float c10 = cb[(size_t)((lt + h1c) * WC + w0c) * CDIM];
    float c11 = cb[(size_t)((lt + h1c) * WC + w1c) * CDIM];
    float vc = (1.f - fh) * ((1.f - fw) * c00 + fw * c01) + fh * ((1.f - fw) * c10 + fw * c11);
    out[idx] = vf + vc + 0.1f * xf;
}

// ==================================================================================
extern "C" void kernel_launch(void* const* d_in, const int* in_sizes, int n_in,
                              void* d_out, int out_size, void* d_ws, size_t ws_size,
                              hipStream_t stream) {
    const float* x     = (const float*)d_in[0];
    const float* ln1_g = (const float*)d_in[1];
    const float* ln1_b = (const float*)d_in[2];
    const float* ln2_g = (const float*)d_in[3];
    const float* ln2_b = (const float*)d_in[4];

    struct BW {
        const float *in_w, *conv_w, *conv_b, *xproj_w, *dt_w, *dt_b, *A_log, *D, *out_w;
    };
    BW fw = { (const float*)d_in[5],  (const float*)d_in[6],  (const float*)d_in[7],
              (const float*)d_in[8],  (const float*)d_in[9],  (const float*)d_in[10],
              (const float*)d_in[11], (const float*)d_in[12], (const float*)d_in[13] };
    BW cw = { (const float*)d_in[14], (const float*)d_in[15], (const float*)d_in[16],
              (const float*)d_in[17], (const float*)d_in[18], (const float*)d_in[19],
              (const float*)d_in[20], (const float*)d_in[21], (const float*)d_in[22] };

    float* ws = (float*)d_ws;
    size_t off = 0;
    auto alloc = [&](size_t n) { float* p = ws + off; off += n; return p; };
    auto alloch = [&](size_t n) {                     // f16 buffer, 16B aligned
        size_t nf = (n + 1) / 2;
        return (_Float16*)alloc(nf + (nf & 3 ? 4 - (nf & 3) : 0));
    };

    // fine-branch buffers
    float*     tokF = alloc((size_t)BSZ * LF * CDIM);
    _Float16*  lnF  = alloch((size_t)BSZ * LF * CDIM);
    float*     xzF  = alloc((size_t)BSZ * LF * 256);
    float*     uF   = alloc((size_t)BSZ * LF * DINNER);
    float*     dtF  = alloc((size_t)BSZ * LF * DINNER);
    float*     BmF  = alloc((size_t)BSZ * LF * DSTATE);
    float*     CmF  = alloc((size_t)BSZ * LF * DSTATE);
    float*     ysF  = alloc((size_t)BSZ * LF * DINNER);
    _Float16*  ygF  = alloch((size_t)BSZ * LF * DINNER);
    float*     yoF  = alloc((size_t)BSZ * LF * CDIM);
    float*     obrF = alloc((size_t)BSZ * LF * CDIM);
    // coarse-branch buffers
    float*     tokC = alloc((size_t)BSZ * LC * CDIM);
    _Float16*  lnC  = alloch((size_t)BSZ * LC * CDIM);
    float*     xzC  = alloc((size_t)BSZ * LC * 256);
    float*     uC   = alloc((size_t)BSZ * LC * DINNER);
    float*     dtC  = alloc((size_t)BSZ * LC * DINNER);
    float*     BmC  = alloc((size_t)BSZ * LC * DSTATE);
    float*     CmC  = alloc((size_t)BSZ * LC * DSTATE);
    float*     ysC  = alloc((size_t)BSZ * LC * DINNER);
    _Float16*  ygC  = alloch((size_t)BSZ * LC * DINNER);
    float*     yoC  = alloc((size_t)BSZ * LC * CDIM);
    float*     obrC = alloc((size_t)BSZ * LC * CDIM);
    // f16 weights (per branch)
    _Float16*  wInF  = alloch((size_t)256 * CDIM);
    _Float16*  wOutF = alloch((size_t)CDIM * DINNER);
    _Float16*  wInC  = alloch((size_t)256 * CDIM);
    _Float16*  wOutC = alloch((size_t)CDIM * DINNER);
    (void)ws_size; (void)n_in; (void)in_sizes; (void)out_size;

    // weight conversion (tiny)
    to_f16_kernel<<<(256 * CDIM + 255) / 256, 256, 0, stream>>>(fw.in_w,  wInF,  256 * CDIM);
    to_f16_kernel<<<(CDIM * DINNER + 255) / 256, 256, 0, stream>>>(fw.out_w, wOutF, CDIM * DINNER);
    to_f16_kernel<<<(256 * CDIM + 255) / 256, 256, 0, stream>>>(cw.in_w,  wInC,  256 * CDIM);
    to_f16_kernel<<<(CDIM * DINNER + 255) / 256, 256, 0, stream>>>(cw.out_w, wOutC, CDIM * DINNER);

    // pooling
    {
        int n = BSZ * LF * CDIM;
        pool_fine_kernel<<<(n + 255) / 256, 256, 0, stream>>>(x, tokF);
        int m = BSZ * LC * CDIM;
        pool_coarse_kernel<<<(m + 255) / 256, 256, 0, stream>>>(tokF, tokC);
    }

    auto run_branch = [&](const float* tok, int L, const BW& W,
                          const _Float16* wIn16, const _Float16* wOut16,
                          _Float16* ln, float* xz, float* u, float* dt,
                          float* Bm, float* Cm, float* ys, _Float16* yg,
                          float* yo, float* obr) {
        int ntok = BSZ * L;
        ln_kernel<<<(ntok + 255) / 256, 256, 0, stream>>>(tok, ln, ln1_g, ln1_b, ntok);
        {   // in-projection: (ntok,64) x (256,64)^T  -- 16x64 strip per wave
            int tiles = (ntok / 16) * (256 / 64);
            gemm_wmma_f16_kernel<<<(tiles + 3) / 4, 128, 0, stream>>>(ln, wIn16, xz, ntok, 256, CDIM);
        }
        {
            int n = ntok * DINNER;
            conv_silu_kernel<<<(n + 255) / 256, 256, 0, stream>>>(xz, W.conv_w, W.conv_b, u, L);
        }
        xdbl_dt_kernel<<<ntok, DINNER, 0, stream>>>(u, W.xproj_w, W.dt_w, W.dt_b, dt, Bm, Cm);
        scan_kernel<<<BSZ * (DINNER / 16), 256, 0, stream>>>(dt, u, Bm, Cm, W.A_log, ys, L);
        {
            int n = ntok * DINNER;
            gate_kernel<<<(n + 255) / 256, 256, 0, stream>>>(ys, u, W.D, xz, yg, n);
        }
        {   // out-projection: (ntok,128) x (64,128)^T
            int tiles = (ntok / 16) * (CDIM / 64);
            gemm_wmma_f16_kernel<<<(tiles + 3) / 4, 128, 0, stream>>>(yg, wOut16, yo, ntok, CDIM, DINNER);
        }
        res_ln_kernel<<<(ntok + 255) / 256, 256, 0, stream>>>(tok, yo, obr, ln2_g, ln2_b, ntok);
    };

    run_branch(tokF, LF, fw, wInF, wOutF, lnF, xzF, uF, dtF, BmF, CmF, ysF, ygF, yoF, obrF);
    run_branch(tokC, LC, cw, wInC, wOutC, lnC, xzC, uC, dtC, BmC, CmC, ysC, ygC, yoC, obrC);

    {
        int n = BSZ * CDIM * TDIM * HF * WF;
        final_kernel<<<(n + 255) / 256, 256, 0, stream>>>(obrF, obrC, tokF, (float*)d_out);
    }
}
